// EquivDiffusion_49658411876808
// MI455X (gfx1250) — matrix operand order repaired and checked by reading, not compiled, not run on hardware
//
#include <hip/hip_runtime.h>

// ---------------------------------------------------------------------------
// ED-HNN style hypergraph diffusion layer, fused for MI455X (gfx1250).
// N=40000 nodes, E=20000 hyperedges, M=640000 incidences, D=128.
// All per-incidence MLP math runs through v_wmma_f32_16x16x32_f16.
// ---------------------------------------------------------------------------

typedef _Float16 h16;
typedef __attribute__((ext_vector_type(16))) _Float16 hv16;
typedef __attribute__((ext_vector_type(8)))  _Float16 hv8;
typedef __attribute__((ext_vector_type(4)))  _Float16 hv4;
typedef __attribute__((ext_vector_type(8)))  float    fv8;

#define D_FEAT 128
#define TILE_M 64          // incidence rows per block
#define BLOCK_T 128        // 4 wave32s
#define E_SEG 20000

// ---------------------------------------------------------------------------
// Per-wave GEMM: 16-row strip (this wave's rows in LDS, f16) times Wt[K x N=128]
// (f16, row-major over output column n, 16B-aligned), bias-initialized f32
// accumulators in WMMA C layout.  accs[nt][r] = elem(row = r + 8*(lane>=16),
// col = nt*16 + lane%16).
// ---------------------------------------------------------------------------
template<int K>
__device__ __forceinline__ void gemm_strip(const h16* __restrict__ aw, int AS,
                                           const h16* __restrict__ wt,
                                           const float* __restrict__ bias,
                                           int lane, fv8 accs[8]) {
  const int nlane = lane & 15;
  const int arow  = nlane;              // local row within this wave's strip
  const int aoff  = (lane >> 4) << 3;   // +8 K offset for lanes 16..31 (A layout)
  const int boff  = (lane >> 4) << 3;   // +8 K offset for lanes 16..31 (B layout)
#pragma unroll
  for (int nt = 0; nt < 8; ++nt) {
    float bv = bias[nt * 16 + nlane];
    fv8 c = {bv, bv, bv, bv, bv, bv, bv, bv};
    const h16* wn = wt + (nt * 16 + nlane) * K + boff;
#pragma unroll
    for (int kc = 0; kc < K / 32; ++kc) {
      // A fragment: lane<16 -> row, K = {kb..kb+7, kb+16..kb+23}; upper lanes kb+=8
      const h16* pa = aw + arow * AS + kc * 32 + aoff;
      hv16 a, b;
      *(hv8*)&a       = *(const hv8*)(pa);
      *((hv8*)&a + 1) = *(const hv8*)(pa + 16);
      // B fragment: lane -> column, K = {kb..kb+7, kb+16..kb+23}; upper lanes kb+=8
      const h16* pb = wn + kc * 32;
      *(hv8*)&b       = *(const hv8*)(pb);
      *((hv8*)&b + 1) = *(const hv8*)(pb + 16);
      c = __builtin_amdgcn_wmma_f32_16x16x32_f16(false, a, false, b,
                                                 (short)0, c, false, false);
    }
    accs[nt] = c;
  }
}

// LayerNorm + ReLU on C-layout accumulators, write f16 rows (stride 136 halves)
// into this wave's own LDS region (local rows 0..15).
__device__ __forceinline__ void ln_relu_store(const fv8 accs[8],
                                              const float* __restrict__ g,
                                              const float* __restrict__ be,
                                              h16* __restrict__ hw, int HS,
                                              int lane) {
  const int nlane = lane & 15;
  const int rup   = (lane >> 4) << 3;   // lanes 16..31 own rows 8..15
  float mu[8], rs[8];
#pragma unroll
  for (int r = 0; r < 8; ++r) {
    float s = 0.f, s2 = 0.f;
#pragma unroll
    for (int t = 0; t < 8; ++t) { float v = accs[t][r]; s += v; s2 += v * v; }
#pragma unroll
    for (int mask = 1; mask <= 8; mask <<= 1) {   // butterfly over 16 col-lanes
      s  += __shfl_xor(s,  mask);
      s2 += __shfl_xor(s2, mask);
    }
    float m   = s  * (1.0f / 128.0f);
    float var = s2 * (1.0f / 128.0f) - m * m;
    mu[r] = m;
    rs[r] = rsqrtf(var + 1e-5f);
  }
#pragma unroll
  for (int t = 0; t < 8; ++t) {
    float gc = g[t * 16 + nlane];
    float bc = be[t * 16 + nlane];
#pragma unroll
    for (int r = 0; r < 8; ++r) {
      float v = (accs[t][r] - mu[r]) * rs[r] * gc + bc;
      v = fmaxf(v, 0.0f);
      hw[(r + rup) * HS + t * 16 + nlane] = (h16)v;
    }
  }
}

// Atomic scatter-add of a wave's 16x128 result strip into outAcc[seg*128 + col].
__device__ __forceinline__ void scatter_add(const fv8 accs[8],
                                            float* __restrict__ outAcc,
                                            const int* __restrict__ segIdx,
                                            int rowBase, int lane) {
  const int nlane = lane & 15;
  const int rup   = (lane >> 4) << 3;
#pragma unroll
  for (int r = 0; r < 8; ++r) {
    int seg = segIdx[rowBase + r + rup];
    float* p = outAcc + (size_t)seg * D_FEAT + nlane;
#pragma unroll
    for (int t = 0; t < 8; ++t) unsafeAtomicAdd(p + t * 16, accs[t][r]);
  }
}

// ---------------------------------------------------------------------------
// Fused per-incidence MLP.  KIN==128: phase 1 (gather x[vertex], MLP1,
// scatter onto edges).  KIN==256: phase 2 (gather [x[vertex] | Xe[edges]],
// MLP2, scatter onto vertices).
// ---------------------------------------------------------------------------
template<int KIN>
__global__ __launch_bounds__(BLOCK_T) void inc_mlp(
    const float* __restrict__ x, const float* __restrict__ xe,
    const long long* __restrict__ vertex, const long long* __restrict__ edges,
    const h16* __restrict__ wtA, const float* __restrict__ bA,
    const float* __restrict__ gamma, const float* __restrict__ beta,
    const h16* __restrict__ wtB, const float* __restrict__ bB,
    float* __restrict__ outAcc, float* __restrict__ cnt) {
  constexpr int AS = KIN + 8;                 // padded LDS row stride (halves)
  __shared__ h16 ldsA[TILE_M * AS];
  __shared__ int segIdx[TILE_M];

  const int tid = threadIdx.x;
  const long long m0 = (long long)blockIdx.x * TILE_M;
  const long long* seg = (KIN == 128) ? edges : vertex;

  // segment indices + per-segment counts (one per incidence row)
  if (tid < TILE_M) {
    int s = (int)seg[m0 + tid];
    segIdx[tid] = s;
    unsafeAtomicAdd(cnt + s, 1.0f);
  }

  // gather input rows into LDS (f16).  2 threads per row.
  {
    const int row = tid >> 1;
    const int sel = tid & 1;
    const long long m = m0 + row;
    const float* src;
    h16* dst = ldsA + row * AS;
    if (KIN == 128) {
      src = x + (size_t)((int)vertex[m]) * D_FEAT + sel * 64;
      dst += sel * 64;
    } else {
      src = sel ? (xe + (size_t)((int)edges[m]) * D_FEAT)
                : (x  + (size_t)((int)vertex[m]) * D_FEAT);
      dst += sel * 128;
    }
    const int NC = (KIN == 128) ? 64 : 128;
#pragma unroll
    for (int c = 0; c < NC; c += 4) {
      float4 f = *(const float4*)(src + c);
      hv4 o = {(h16)f.x, (h16)f.y, (h16)f.z, (h16)f.w};
      *(hv4*)(dst + c) = o;
    }
  }
  __syncthreads();

  const int lane = tid & 31;
  const int w    = tid >> 5;                  // wave id, strip rows 16w..16w+15
  h16* aw = ldsA + w * 16 * AS;               // this wave's LDS region

  fv8 accs[8];
  gemm_strip<KIN>(aw, AS, wtA, bA, lane, accs);           // Linear A (+bias)
  ln_relu_store(accs, gamma, beta, aw, 136, lane);        // LN + ReLU -> LDS
  // (per-wave LDS is in-order; region only aliases this wave's own A rows)
  gemm_strip<128>(aw, 136, wtB, bB, lane, accs);          // Linear B (+bias)
  scatter_add(accs, outAcc, segIdx, w * 16, lane);        // segment sums
}

// ---------------------------------------------------------------------------
// Small helper kernels
// ---------------------------------------------------------------------------
__global__ void prep_w(const float* __restrict__ w11, const float* __restrict__ w12,
                       const float* __restrict__ w21, const float* __restrict__ w22,
                       h16* __restrict__ t11, h16* __restrict__ t12,
                       h16* __restrict__ t21, h16* __restrict__ t22) {
  int i = blockIdx.x * 256 + threadIdx.x;     // transpose + f32->f16
  if (i < 16384)      { int n = i >> 7, k = i & 127;              t11[i] = (h16)w11[k * 128 + n]; }
  else if (i < 32768) { int j = i - 16384; int n = j >> 7, k = j & 127; t12[j] = (h16)w12[k * 128 + n]; }
  else if (i < 65536) { int j = i - 32768; int n = j >> 8, k = j & 255; t21[j] = (h16)w21[k * 128 + n]; }
  else if (i < 81920) { int j = i - 65536; int n = j >> 7, k = j & 127; t22[j] = (h16)w22[k * 128 + n]; }
}

__global__ void seg_div(float* __restrict__ acc, const float* __restrict__ cnt, int n) {
  int i = blockIdx.x * 256 + threadIdx.x;
  if (i < n) acc[i] /= fmaxf(cnt[i >> 7], 1.0f);
}

__global__ void finalize(const float* __restrict__ xvAcc, const float* __restrict__ cntV,
                         const float* __restrict__ x, float* __restrict__ out, int n) {
  int i = blockIdx.x * 256 + threadIdx.x;
  if (i < n) {
    float xv = xvAcc[i] / fmaxf(cntV[i >> 7], 1.0f);
    out[i] = 0.9f * xv + 0.1f * x[i];
  }
}

// ---------------------------------------------------------------------------
extern "C" void kernel_launch(void* const* d_in, const int* in_sizes, int n_in,
                              void* d_out, int out_size, void* d_ws, size_t ws_size,
                              hipStream_t stream) {
  const float* x   = (const float*)d_in[0];
  const float* w11 = (const float*)d_in[1];
  const float* b11 = (const float*)d_in[2];
  const float* g1  = (const float*)d_in[3];
  const float* be1 = (const float*)d_in[4];
  const float* w12 = (const float*)d_in[5];
  const float* b12 = (const float*)d_in[6];
  const float* w21 = (const float*)d_in[7];
  const float* b21 = (const float*)d_in[8];
  const float* g2  = (const float*)d_in[9];
  const float* be2 = (const float*)d_in[10];
  const float* w22 = (const float*)d_in[11];
  const float* b22 = (const float*)d_in[12];
  const long long* ei = (const long long*)d_in[13];

  const int M = in_sizes[13] / 2;
  const int N = in_sizes[0] / D_FEAT;
  const int E = E_SEG;
  const long long* vtx = ei;
  const long long* edg = ei + M;

  char* ws = (char*)d_ws;
  size_t off = 0;
  auto take = [&](size_t bytes) {
    size_t o = off;
    off += (bytes + 255) & ~(size_t)255;
    return o;
  };
  h16* wt11 = (h16*)(ws + take((size_t)128 * 128 * 2));
  h16* wt12 = (h16*)(ws + take((size_t)128 * 128 * 2));
  h16* wt21 = (h16*)(ws + take((size_t)256 * 128 * 2));
  h16* wt22 = (h16*)(ws + take((size_t)128 * 128 * 2));
  size_t zeroStart = off;
  float* Xe   = (float*)(ws + take((size_t)E * D_FEAT * 4));
  float* cntE = (float*)(ws + take((size_t)E * 4));
  float* Xv   = (float*)(ws + take((size_t)N * D_FEAT * 4));
  float* cntV = (float*)(ws + take((size_t)N * 4));

  // zero segment accumulators (graph-capture safe)
  hipMemsetAsync(ws + zeroStart, 0, off - zeroStart, stream);

  // weights -> f16, transposed to [nout][K]
  prep_w<<<320, 256, 0, stream>>>(w11, w12, w21, w22, wt11, wt12, wt21, wt22);

  // phase 1: Xe_acc = segment_sum(MLP1(x[vertex]), edges)
  inc_mlp<128><<<M / TILE_M, BLOCK_T, 0, stream>>>(
      x, (const float*)nullptr, vtx, edg, wt11, b11, g1, be1, wt12, b12, Xe, cntE);

  // Xe = Xe_acc / max(cnt, 1)
  seg_div<<<(E * D_FEAT + 255) / 256, 256, 0, stream>>>(Xe, cntE, E * D_FEAT);

  // phase 2: Xv_acc = segment_sum(MLP2([x[vertex] | Xe[edges]]), vertex)
  inc_mlp<256><<<M / TILE_M, BLOCK_T, 0, stream>>>(
      x, Xe, vtx, edg, wt21, b21, g2, be2, wt22, b22, Xv, cntV);

  // out = 0.9 * Xv_acc/cnt + 0.1 * x
  finalize<<<(N * D_FEAT + 255) / 256, 256, 0, stream>>>(
      Xv, cntV, x, (float*)d_out, N * D_FEAT);
}
